// SelfAttention_38311108280946
// MI455X (gfx1250) — matrix operand order repaired and checked by reading, not compiled, not run on hardware
//
#include <hip/hip_runtime.h>
#include <hip/hip_bf16.h>

#define BB 32
#define SS 2048
#define DD 1024
#define AA 256
#define XP 1032   // LDS row pitch in ushorts (516 dwords/row -> bank stride 4, conflict-free)

typedef __attribute__((ext_vector_type(16))) __bf16          v16bf;
typedef __attribute__((ext_vector_type(16))) unsigned short  v16u;
typedef __attribute__((ext_vector_type(8)))  unsigned short  u16x8;
typedef __attribute__((ext_vector_type(8)))  float           v8f;

__device__ __forceinline__ unsigned short f2bf(float f) {
    unsigned int ub = __float_as_uint(f);
    ub += 0x7FFFu + ((ub >> 16) & 1u);   // round-to-nearest-even
    return (unsigned short)(ub >> 16);
}

// Branchless tanh: 1 - 2/(e^{2x}+1). Correct limits (+1/-1) as exp saturates,
// no EXEC-mask divergence (keeps the wave WMMA-legal throughout).
__device__ __forceinline__ float fast_tanh(float v) {
    const float e = __expf(2.0f * v);                 // v_exp_f32 path
    return 1.0f - 2.0f * __builtin_amdgcn_rcpf(e + 1.0f);
}

// ---------------------------------------------------------------------------
// Kernel 1: logits[b, s] = tanh(x[b,s,:] @ W + b) @ u
// One wave per 16-row x tile. v_wmma_f32_16x16x32_bf16, 2 N-tiles register-
// blocked per pass (A operand shared between the two WMMAs).
// ---------------------------------------------------------------------------
__global__ void __launch_bounds__(32)
attn_logits_kernel(const float* __restrict__ x,
                   const float* __restrict__ Wm,
                   const float* __restrict__ bias,
                   const float* __restrict__ u,
                   float* __restrict__ logits)
{
    __shared__ __align__(16) unsigned short xs [16 * XP];  // x tile bf16  [m][k]   (33 KB)
    __shared__ __align__(16) unsigned short wsT[32 * XP];  // W slab bf16  [c][k]   (66 KB)

    const int lane = threadIdx.x;                 // 0..31
    const int b    = blockIdx.x >> 7;             // S/16 = 128 tiles per batch
    const int s0   = (blockIdx.x & 127) << 4;

    // Stage the full 16 x 1024 x-tile once (bf16, coalesced)
    const float* xbase = x + ((size_t)b * SS + s0) * DD;
    for (int idx = lane; idx < 16 * DD; idx += 32) {
        const int row = idx >> 10;
        const int col = idx & (DD - 1);
        xs[row * XP + col] = f2bf(xbase[(size_t)row * DD + col]);
    }

    const int half = lane >> 4;        // 0: lanes 0-15, 1: lanes 16-31
    const int mcol = lane & 15;        // M row group for A, N column for B/C
    const int aoff = half ? 16 : 0;    // A: lanes>=16 hold K 16..31 (matches B)
    const int boff = half ? 16 : 0;    // B: lanes>=16 hold K 16..31

    const unsigned short* arow  = xs  + mcol * XP;
    const unsigned short* brow0 = wsT + mcol * XP;          // N-tile 0 of the pair
    const unsigned short* brow1 = wsT + (16 + mcol) * XP;   // N-tile 1 of the pair

    float plog[8];
#pragma unroll
    for (int r = 0; r < 8; ++r) plog[r] = 0.0f;

    for (int p = 0; p < AA / 32; ++p) {            // 8 passes of 32 columns
        __syncthreads();
        // Stage W[:, 32p : 32p+32] transposed -> wsT[c][k] (contiguous B reads)
        for (int idx = lane; idx < 32 * DD; idx += 32) {
            const int k = idx >> 5;
            const int c = idx & 31;
            wsT[c * XP + k] = f2bf(Wm[(size_t)k * AA + p * 32 + c]);
        }
        __syncthreads();

        v8f acc0 = {};
        v8f acc1 = {};
#pragma unroll 4
        for (int k0 = 0; k0 < DD; k0 += 32) {
            // A operand: shared by both WMMAs (two contiguous 8-element runs)
            u16x8 a0 = *(const u16x8*)(arow + k0 + aoff);
            u16x8 a1 = *(const u16x8*)(arow + k0 + aoff + 8);
            // B operands for the two N-tiles (contiguous K runs)
            u16x8 p0 = *(const u16x8*)(brow0 + k0 + boff);
            u16x8 p1 = *(const u16x8*)(brow0 + k0 + boff + 8);
            u16x8 q0 = *(const u16x8*)(brow1 + k0 + boff);
            u16x8 q1 = *(const u16x8*)(brow1 + k0 + boff + 8);
            v16u au, bu, cu;
#pragma unroll
            for (int i = 0; i < 8; ++i) {
                au[i] = a0[i]; au[8 + i] = a1[i];
                bu[i] = p0[i]; bu[8 + i] = p1[i];
                cu[i] = q0[i]; cu[8 + i] = q1[i];
            }
            const v16bf av = __builtin_bit_cast(v16bf, au);
            acc0 = __builtin_amdgcn_wmma_f32_16x16x32_bf16(
                       false, av, false, __builtin_bit_cast(v16bf, bu),
                       (short)0, acc0, false, false);
            acc1 = __builtin_amdgcn_wmma_f32_16x16x32_bf16(
                       false, av, false, __builtin_bit_cast(v16bf, cu),
                       (short)0, acc1, false, false);
        }

        // tanh + dot with u for both 16-column slabs.
        // C layout: VGPR r -> M = r + 8*half, N = mcol
        const int c0 = p * 32 + mcol;
        const int c1 = c0 + 16;
        const float uv0 = u[c0],    uv1 = u[c1];
        const float bv0 = bias[c0], bv1 = bias[c1];
#pragma unroll
        for (int r = 0; r < 8; ++r) {
            plog[r] += fast_tanh(acc0[r] + bv0) * uv0;
            plog[r] += fast_tanh(acc1[r] + bv1) * uv1;
        }
    }

    // Row-sum across the 16 lanes of each half
#pragma unroll
    for (int r = 0; r < 8; ++r) {
        float v = plog[r];
        v += __shfl_xor(v, 1, 32);
        v += __shfl_xor(v, 2, 32);
        v += __shfl_xor(v, 4, 32);
        v += __shfl_xor(v, 8, 32);
        plog[r] = v;
    }
    if (mcol == 0) {
        const int mbase = s0 + (half ? 8 : 0);
#pragma unroll
        for (int r = 0; r < 8; ++r)
            logits[(size_t)b * SS + mbase + r] = plog[r];
    }
}

// ---------------------------------------------------------------------------
// Kernel 2: masked softmax over S, then out[b,:] = sum_s w[s] * x[b,s,:]
// One 256-thread block per batch.
// ---------------------------------------------------------------------------
__global__ void __launch_bounds__(256)
attn_pool_kernel(const float* __restrict__ x,
                 const int*   __restrict__ mask,
                 const float* __restrict__ logits,
                 float* __restrict__ out)
{
    __shared__ float w[SS];      // 8 KB
    __shared__ float red[256];

    const int b = blockIdx.x;
    const int t = threadIdx.x;

    const float NEG_INF = -__builtin_inff();
    float lmax = NEG_INF;
    for (int s = t; s < SS; s += 256) {
        float v = logits[(size_t)b * SS + s];
        if (mask[(size_t)b * SS + s] == 0) v = NEG_INF;
        w[s] = v;
        lmax = fmaxf(lmax, v);
    }
    red[t] = lmax;
    __syncthreads();
    for (int st = 128; st > 0; st >>= 1) {
        if (t < st) red[t] = fmaxf(red[t], red[t + st]);
        __syncthreads();
    }
    const float gmax = red[0];
    __syncthreads();

    float lsum = 0.0f;
    for (int s = t; s < SS; s += 256) {
        const float e = __expf(w[s] - gmax);
        w[s] = e;
        lsum += e;
    }
    red[t] = lsum;
    __syncthreads();
    for (int st = 128; st > 0; st >>= 1) {
        if (t < st) red[t] += red[t + st];
        __syncthreads();
    }
    const float inv = 1.0f / red[0];
    __syncthreads();

    // Weighted sum: thread t owns columns t, t+256, t+512, t+768 (coalesced)
    float acc0 = 0.f, acc1 = 0.f, acc2 = 0.f, acc3 = 0.f;
    const float* xb = x + (size_t)b * SS * DD + t;
    for (int s = 0; s < SS; ++s) {
        const float ws = w[s];
        const float* xr = xb + (size_t)s * DD;
        acc0 += ws * xr[0];
        acc1 += ws * xr[256];
        acc2 += ws * xr[512];
        acc3 += ws * xr[768];
    }
    float* ob = out + (size_t)b * DD + t;
    ob[0]   = acc0 * inv;
    ob[256] = acc1 * inv;
    ob[512] = acc2 * inv;
    ob[768] = acc3 * inv;
}

extern "C" void kernel_launch(void* const* d_in, const int* in_sizes, int n_in,
                              void* d_out, int out_size, void* d_ws, size_t ws_size,
                              hipStream_t stream)
{
    const float* x    = (const float*)d_in[0];
    const int*   mask = (const int*)  d_in[1];
    const float* Wm   = (const float*)d_in[2];
    const float* bias = (const float*)d_in[3];
    const float* u    = (const float*)d_in[4];
    float* out    = (float*)d_out;
    float* logits = (float*)d_ws;     // [B, S] fp32 = 256 KB scratch

    attn_logits_kernel<<<dim3(BB * (SS / 16)), dim3(32), 0, stream>>>(
        x, Wm, bias, u, logits);
    attn_pool_kernel<<<dim3(BB), dim3(256), 0, stream>>>(
        x, mask, logits, out);
}
